// cross_attention_75831942578342
// MI455X (gfx1250) — matrix operand order repaired and checked by reading, not compiled, not run on hardware
//
#include <hip/hip_runtime.h>
#include <hip/hip_bf16.h>

typedef _Float16 v16h __attribute__((ext_vector_type(16)));
typedef _Float16 v8h  __attribute__((ext_vector_type(8)));
typedef float    v8f  __attribute__((ext_vector_type(8)));

#define B_  64
#define LS_ 512
#define LT_ 256
#define D_  256
#define H_  8
#define DH_ 32
#define LN_EPS  1e-5f
#define ATT_EPS 1e-6f

enum { EP_BIAS = 1, EP_RES = 2, EP_RELU = 4, EP_ELU1 = 8, EP_MASK = 16 };

static __device__ __forceinline__ v16h make16(v8h lo, v8h hi) {
  v16h r;
#pragma unroll
  for (int i = 0; i < 8; ++i) { r[i] = lo[i]; r[i + 8] = hi[i]; }
  return r;
}

static __device__ __forceinline__ v8f wmma16(v16h a, v16h b, v8f c) {
  return __builtin_amdgcn_wmma_f32_16x16x32_f16(false, a, false, b, (short)0, c,
                                                false, false);
}

// CDNA5 async copy: global -> LDS, 16B per lane, tracked by ASYNCcnt.
static __device__ __forceinline__ void async_cp16(const _Float16* g,
                                                  _Float16* l) {
  unsigned lofs = (unsigned)(unsigned long long)(uintptr_t)l;  // LDS byte addr
  asm volatile("global_load_async_to_lds_b128 %0, %1, off"
               :
               : "v"(lofs), "v"((unsigned long long)(uintptr_t)g)
               : "memory");
}

static __device__ __forceinline__ void wait_async0() {
  asm volatile("s_wait_asynccnt 0x0" ::: "memory");
}

// ---------------------------------------------------------------------------
// Generic tiled WMMA GEMM: C[M,N] = epilogue(A[M,K] @ W[K,N])
// A: f16 row-major (lda), Wt: f16 [N][K] row-major (ldw) == W transposed.
// Block: 256 threads = 8 waves (4 along M x 2 along N), tile 128x128, BK=32.
// Ping-pong LDS buffers; next tile staged via async-to-LDS while WMMAs run.
// ---------------------------------------------------------------------------
template <int EP>
__global__ __launch_bounds__(256) void gemm_wmma_kernel(
    const _Float16* __restrict__ A, const _Float16* __restrict__ Wt,
    int M, int N, int K, int lda, int ldw, int ldc,
    float* __restrict__ out32, _Float16* __restrict__ out16,
    const float* __restrict__ bias, const float* __restrict__ residual,
    const unsigned char* __restrict__ rowmask, float scale) {
  __shared__ _Float16 As[2][128 * 40];
  __shared__ _Float16 Bs[2][128 * 40];

  const int tid = threadIdx.x;
  const int lane = tid & 31, wave = tid >> 5;
  const int bm = blockIdx.y * 128, bn = blockIdx.x * 128;
  const int wm = (wave & 3) * 32, wn = (wave >> 2) * 64;
  const int hi = lane >> 4, r = lane & 15;

  v8f acc[2][4];
#pragma unroll
  for (int mt = 0; mt < 2; ++mt)
#pragma unroll
    for (int nt = 0; nt < 4; ++nt)
#pragma unroll
      for (int i = 0; i < 8; ++i) acc[mt][nt][i] = 0.0f;

  auto stage = [&](int kb, int sel) {
#pragma unroll
    for (int rep = 0; rep < 2; ++rep) {
      int c = tid + rep * 256;
      int row = c >> 2, kp = (c & 3) << 3;
      async_cp16(A + (size_t)(bm + row) * lda + kb + kp,
                 &As[sel][row * 40 + kp]);
      async_cp16(Wt + (size_t)(bn + row) * ldw + kb + kp,
                 &Bs[sel][row * 40 + kp]);
    }
  };

  stage(0, 0);
  int cur = 0;
  for (int kb = 0; kb < K; kb += 32) {
    wait_async0();        // buffer `cur` fully written by this wave's asyncs
    __syncthreads();      // ... and by every other wave's
    if (kb + 32 < K) stage(kb + 32, cur ^ 1);

    v16h af[2], bf[4];
#pragma unroll
    for (int mt = 0; mt < 2; ++mt) {
      const _Float16* p = &As[cur][(wm + mt * 16 + r) * 40 + hi * 16];
      af[mt] = make16(*(const v8h*)p, *(const v8h*)(p + 8));
    }
#pragma unroll
    for (int nt = 0; nt < 4; ++nt) {
      const _Float16* p = &Bs[cur][(wn + nt * 16 + r) * 40 + hi * 16];
      bf[nt] = make16(*(const v8h*)p, *(const v8h*)(p + 8));
    }
#pragma unroll
    for (int mt = 0; mt < 2; ++mt)
#pragma unroll
      for (int nt = 0; nt < 4; ++nt)
        acc[mt][nt] = wmma16(af[mt], bf[nt], acc[mt][nt]);
    cur ^= 1;
  }

#pragma unroll
  for (int mt = 0; mt < 2; ++mt) {
#pragma unroll
    for (int nt = 0; nt < 4; ++nt) {
#pragma unroll
      for (int i = 0; i < 8; ++i) {
        int row = bm + wm + mt * 16 + ((lane < 16) ? i : i + 8);
        int col = bn + wn + nt * 16 + r;
        float y = acc[mt][nt][i];
        if constexpr (EP & EP_BIAS) y += bias[col];
        if constexpr (EP & EP_RES)  y += residual[(size_t)row * ldc + col];
        if constexpr (EP & EP_ELU1) y = (y > 0.0f) ? (y + 1.0f) : __expf(y);
        if constexpr (EP & EP_RELU) y = fmaxf(y, 0.0f);
        if constexpr (EP & EP_MASK) y = rowmask[row] ? y : 0.0f;
        y *= scale;
        size_t o = (size_t)row * ldc + col;
        if (out16) out16[o] = (_Float16)y;
        if (out32) out32[o] = y;
      }
    }
  }
}

// ---------------------------------------------------------------------------
// KV[b,h] = K_head^T @ V_head  (DHxDH, K-dim = LT). One block (4 waves) per
// (b,h). Output stored transposed+f16 as KVt[b,h][v][d] (B-operand layout).
// ---------------------------------------------------------------------------
__global__ __launch_bounds__(128) void kv_kernel(
    const _Float16* __restrict__ Kf, const _Float16* __restrict__ Vf,
    _Float16* __restrict__ KVt) {
  __shared__ _Float16 Kt[4][32 * 40];
  __shared__ _Float16 Vt[4][32 * 40];
  __shared__ float Cred[4][32 * 32];

  const int bh = blockIdx.x;            // b*H + h
  const int b = bh >> 3, h = bh & 7;
  const int tid = threadIdx.x, lane = tid & 31, wave = tid >> 5;
  const int hi = lane >> 4, r = lane & 15;
  const size_t base = (size_t)b * LT_ * D_ + h * DH_;

  v8f acc[2][2];
#pragma unroll
  for (int mt = 0; mt < 2; ++mt)
#pragma unroll
    for (int nt = 0; nt < 2; ++nt)
#pragma unroll
      for (int i = 0; i < 8; ++i) acc[mt][nt][i] = 0.0f;

  for (int it = 0; it < 2; ++it) {
    int s0 = (wave * 2 + it) * 32;
    const _Float16* kp = Kf + base + (size_t)(s0 + lane) * D_;
    const _Float16* vp = Vf + base + (size_t)(s0 + lane) * D_;
    v8h kk[4], vv[4];
#pragma unroll
    for (int j = 0; j < 4; ++j) {
      kk[j] = *(const v8h*)(kp + j * 8);
      vv[j] = *(const v8h*)(vp + j * 8);
    }
#pragma unroll
    for (int j = 0; j < 4; ++j)
#pragma unroll
      for (int e = 0; e < 8; ++e) {
        Kt[wave][(j * 8 + e) * 40 + lane] = kk[j][e];  // [d][s]
        Vt[wave][(j * 8 + e) * 40 + lane] = vv[j][e];  // [v][s]
      }
    v16h af[2], bf[2];
#pragma unroll
    for (int mt = 0; mt < 2; ++mt) {
      const _Float16* p = &Kt[wave][(mt * 16 + r) * 40 + hi * 16];
      af[mt] = make16(*(const v8h*)p, *(const v8h*)(p + 8));
    }
#pragma unroll
    for (int nt = 0; nt < 2; ++nt) {
      const _Float16* p = &Vt[wave][(nt * 16 + r) * 40 + hi * 16];
      bf[nt] = make16(*(const v8h*)p, *(const v8h*)(p + 8));
    }
#pragma unroll
    for (int mt = 0; mt < 2; ++mt)
#pragma unroll
      for (int nt = 0; nt < 2; ++nt)
        acc[mt][nt] = wmma16(af[mt], bf[nt], acc[mt][nt]);
  }

#pragma unroll
  for (int mt = 0; mt < 2; ++mt)
#pragma unroll
    for (int nt = 0; nt < 2; ++nt)
#pragma unroll
      for (int i = 0; i < 8; ++i) {
        int d = mt * 16 + ((lane < 16) ? i : i + 8);
        int v = nt * 16 + r;
        Cred[wave][d * 32 + v] = acc[mt][nt][i];
      }
  __syncthreads();
#pragma unroll
  for (int e = 0; e < 8; ++e) {
    int idx = tid + e * 128;  // = d*32 + v
    int d = idx >> 5, v = idx & 31;
    float s = Cred[0][idx] + Cred[1][idx] + Cred[2][idx] + Cred[3][idx];
    KVt[(size_t)bh * 1024 + v * 32 + d] = (_Float16)s;
  }
}

// ---------------------------------------------------------------------------
// msg[b,l,h,v] = (Q[b,l,h,:] @ KV[b,h]) * Z[b,l,h]  (Z already includes *LT)
// One wave per 32 rows of one (b,h); K=32 -> single WMMA k-step.
// ---------------------------------------------------------------------------
__global__ __launch_bounds__(256) void attn_msg_kernel(
    const _Float16* __restrict__ Qf, const _Float16* __restrict__ KVt,
    const float* __restrict__ Z, _Float16* __restrict__ msg) {
  const int g = blockIdx.x * 8 + (threadIdx.x >> 5);
  const int lane = threadIdx.x & 31;
  const int rt = g & 15, bh = g >> 4;
  const int b = bh >> 3, h = bh & 7;
  const int l0 = rt * 32;
  const int hi = lane >> 4, r = lane & 15;

  const size_t qbase = (size_t)(b * LS_ + l0) * D_ + h * DH_;
  v16h af[2], bf[2];
#pragma unroll
  for (int mt = 0; mt < 2; ++mt) {
    const _Float16* p = Qf + qbase + (size_t)(mt * 16 + r) * D_ + hi * 16;
    af[mt] = make16(*(const v8h*)p, *(const v8h*)(p + 8));
  }
  const _Float16* kvb = KVt + (size_t)bh * 1024;
#pragma unroll
  for (int nt = 0; nt < 2; ++nt) {
    const _Float16* p = kvb + (nt * 16 + r) * 32 + hi * 16;
    bf[nt] = make16(*(const v8h*)p, *(const v8h*)(p + 8));
  }

  v8f acc[2][2];
#pragma unroll
  for (int mt = 0; mt < 2; ++mt)
#pragma unroll
    for (int nt = 0; nt < 2; ++nt) {
#pragma unroll
      for (int i = 0; i < 8; ++i) acc[mt][nt][i] = 0.0f;
      acc[mt][nt] = wmma16(af[mt], bf[nt], acc[mt][nt]);
    }

#pragma unroll
  for (int mt = 0; mt < 2; ++mt)
#pragma unroll
    for (int nt = 0; nt < 2; ++nt)
#pragma unroll
      for (int i = 0; i < 8; ++i) {
        int l = l0 + mt * 16 + ((lane < 16) ? i : i + 8);
        int v = nt * 16 + r;
        float z = Z[((size_t)b * LS_ + l) * H_ + h];
        msg[((size_t)b * LS_ + l) * D_ + h * DH_ + v] =
            (_Float16)(acc[mt][nt][i] * z);
      }
}

// --------------------------- small helper kernels ---------------------------
__global__ void transpose_w_kernel(const float* __restrict__ W,
                                   _Float16* __restrict__ Wt, int K, int N) {
  int idx = blockIdx.x * blockDim.x + threadIdx.x;
  if (idx >= K * N) return;
  int n = idx / K, k = idx % K;
  Wt[(size_t)n * K + k] = (_Float16)W[(size_t)k * N + n];
}

__global__ void convert_f16_kernel(const float* __restrict__ X,
                                   _Float16* __restrict__ Y, int n) {
  int idx = blockIdx.x * blockDim.x + threadIdx.x;
  if (idx < n) Y[idx] = (_Float16)X[idx];
}

__global__ void pos_hidden_kernel(const float* __restrict__ coors,
                                  const float* __restrict__ w1,
                                  const float* __restrict__ b1,
                                  _Float16* __restrict__ out, int M) {
  int idx = blockIdx.x * blockDim.x + threadIdx.x;
  if (idx >= M * D_) return;
  int m = idx / D_, n = idx % D_;
  float y = coors[m * 3 + 0] * w1[n] + coors[m * 3 + 1] * w1[D_ + n] +
            coors[m * 3 + 2] * w1[2 * D_ + n] + b1[n];
  out[idx] = (_Float16)fmaxf(y, 0.0f);
}

__global__ void ksum_kernel(const _Float16* __restrict__ Kf,
                            float* __restrict__ Ksum) {
  int idx = blockIdx.x * blockDim.x + threadIdx.x;  // B_*D_ threads
  if (idx >= B_ * D_) return;
  int b = idx / D_, d = idx % D_;
  const _Float16* p = Kf + (size_t)b * LT_ * D_ + d;
  float s = 0.0f;
  for (int t = 0; t < LT_; ++t) s += (float)p[(size_t)t * D_];
  Ksum[idx] = s;  // [b][h*DH+dh]
}

__global__ void z_kernel(const _Float16* __restrict__ Qf,
                         const float* __restrict__ Ksum,
                         float* __restrict__ Z) {
  int idx = blockIdx.x * blockDim.x + threadIdx.x;  // B_*LS_*H_ threads
  if (idx >= B_ * LS_ * H_) return;
  int bl = idx / H_, h = idx % H_;
  const _Float16* q = Qf + (size_t)bl * D_ + h * DH_;
  const float* ks = Ksum + (size_t)(bl / LS_) * D_ + h * DH_;
  float s = 0.0f;
#pragma unroll
  for (int d = 0; d < DH_; ++d) s += (float)q[d] * ks[d];
  Z[idx] = (float)LT_ / (s + ATT_EPS);
}

// LN over D=256 per row + write concat [f16(search_feat) | ln(msg)] f16.
__global__ __launch_bounds__(256) void ln1_cat_kernel(
    const float* __restrict__ X, const float* __restrict__ sfeat,
    const float* __restrict__ g, const float* __restrict__ bta,
    _Float16* __restrict__ cat) {
  int row = blockIdx.x * 8 + (threadIdx.x >> 5);
  int lane = threadIdx.x & 31;
  const float* x = X + (size_t)row * D_;
  float v[8], s = 0.0f;
#pragma unroll
  for (int e = 0; e < 8; ++e) { v[e] = x[lane * 8 + e]; s += v[e]; }
#pragma unroll
  for (int off = 16; off > 0; off >>= 1) s += __shfl_xor(s, off, 32);
  float mu = s * (1.0f / D_);
  float q = 0.0f;
#pragma unroll
  for (int e = 0; e < 8; ++e) { float d = v[e] - mu; q += d * d; }
#pragma unroll
  for (int off = 16; off > 0; off >>= 1) q += __shfl_xor(q, off, 32);
  float inv = rsqrtf(q * (1.0f / D_) + LN_EPS);
  _Float16* crow = cat + (size_t)row * (2 * D_);
  const float* sf = sfeat + (size_t)row * D_;
#pragma unroll
  for (int e = 0; e < 8; ++e) {
    int c = lane * 8 + e;
    crow[c] = (_Float16)sf[c];
    crow[D_ + c] = (_Float16)((v[e] - mu) * inv * g[c] + bta[c]);
  }
}

__global__ __launch_bounds__(256) void ln2_res_kernel(
    const float* __restrict__ X, const float* __restrict__ sfeat,
    const float* __restrict__ g, const float* __restrict__ bta,
    float* __restrict__ out) {
  int row = blockIdx.x * 8 + (threadIdx.x >> 5);
  int lane = threadIdx.x & 31;
  const float* x = X + (size_t)row * D_;
  float v[8], s = 0.0f;
#pragma unroll
  for (int e = 0; e < 8; ++e) { v[e] = x[lane * 8 + e]; s += v[e]; }
#pragma unroll
  for (int off = 16; off > 0; off >>= 1) s += __shfl_xor(s, off, 32);
  float mu = s * (1.0f / D_);
  float q = 0.0f;
#pragma unroll
  for (int e = 0; e < 8; ++e) { float d = v[e] - mu; q += d * d; }
#pragma unroll
  for (int off = 16; off > 0; off >>= 1) q += __shfl_xor(q, off, 32);
  float inv = rsqrtf(q * (1.0f / D_) + LN_EPS);
  const float* sf = sfeat + (size_t)row * D_;
  float* orow = out + (size_t)row * D_;
#pragma unroll
  for (int e = 0; e < 8; ++e) {
    int c = lane * 8 + e;
    orow[c] = sf[c] + ((v[e] - mu) * inv * g[c] + bta[c]);
  }
}

// ---------------------------------------------------------------------------
extern "C" void kernel_launch(void* const* d_in, const int* in_sizes, int n_in,
                              void* d_out, int out_size, void* d_ws,
                              size_t ws_size, hipStream_t stream) {
  const float* search_feat  = (const float*)d_in[0];
  const float* search_coors = (const float*)d_in[1];
  const unsigned char* smask = (const unsigned char*)d_in[2];
  const float* template_feat = (const float*)d_in[3];
  const unsigned char* tmask = (const unsigned char*)d_in[5];
  const float* pos_w1 = (const float*)d_in[6];
  const float* pos_b1 = (const float*)d_in[7];
  const float* pos_w2 = (const float*)d_in[8];
  const float* pos_b2 = (const float*)d_in[9];
  const float* wq = (const float*)d_in[10];
  const float* wk = (const float*)d_in[11];
  const float* wv = (const float*)d_in[12];
  const float* w_merge = (const float*)d_in[13];
  const float* mlp_w1 = (const float*)d_in[14];
  const float* mlp_w2 = (const float*)d_in[15];
  const float* ln1_g = (const float*)d_in[16];
  const float* ln1_b = (const float*)d_in[17];
  const float* ln2_g = (const float*)d_in[18];
  const float* ln2_b = (const float*)d_in[19];
  float* outp = (float*)d_out;

  const int MS = B_ * LS_;  // 32768
  const int MT = B_ * LT_;  // 16384

  char* w = (char*)d_ws;
  size_t off = 0;
  auto alloc = [&](size_t bytes) -> void* {
    void* p = (void*)(w + off);
    off = (off + bytes + 255) & ~(size_t)255;
    return p;
  };
  _Float16* wt_pw2 = (_Float16*)alloc((size_t)D_ * D_ * 2);
  _Float16* wt_q   = (_Float16*)alloc((size_t)D_ * D_ * 2);
  _Float16* wt_k   = (_Float16*)alloc((size_t)D_ * D_ * 2);
  _Float16* wt_v   = (_Float16*)alloc((size_t)D_ * D_ * 2);
  _Float16* wt_m   = (_Float16*)alloc((size_t)D_ * D_ * 2);
  _Float16* wt_m1  = (_Float16*)alloc((size_t)(2 * D_) * (2 * D_) * 2);
  _Float16* wt_m2  = (_Float16*)alloc((size_t)(2 * D_) * D_ * 2);
  _Float16* tmpl_h = (_Float16*)alloc((size_t)MT * D_ * 2);
  _Float16* hid0_h = (_Float16*)alloc((size_t)MS * D_ * 2);
  _Float16* sfp_h  = (_Float16*)alloc((size_t)MS * D_ * 2);
  _Float16* Qf     = (_Float16*)alloc((size_t)MS * D_ * 2);
  _Float16* Kf     = (_Float16*)alloc((size_t)MT * D_ * 2);
  _Float16* Vf     = (_Float16*)alloc((size_t)MT * D_ * 2);
  _Float16* KVt    = (_Float16*)alloc((size_t)B_ * H_ * DH_ * DH_ * 2);
  float*    Ksum   = (float*)alloc((size_t)B_ * D_ * 4);
  float*    Zbuf   = (float*)alloc((size_t)MS * H_ * 4);
  _Float16* msg_h  = (_Float16*)alloc((size_t)MS * D_ * 2);
  float*    msgM   = (float*)alloc((size_t)MS * D_ * 4);
  _Float16* cat_h  = (_Float16*)alloc((size_t)MS * 2 * D_ * 2);
  _Float16* hid_h  = (_Float16*)alloc((size_t)MS * 2 * D_ * 2);
  float*    mlpo   = (float*)alloc((size_t)MS * D_ * 4);
  (void)ws_size; (void)n_in; (void)in_sizes; (void)out_size;

  dim3 blk(256);
  auto gemm = [&](int ep, const _Float16* A, const _Float16* Wt, int M, int N,
                  int K, int lda, int ldw, int ldc, float* o32, _Float16* o16,
                  const float* bias, const float* res,
                  const unsigned char* mask, float scale) {
    dim3 grid(N / 128, M / 128);
    switch (ep) {
      case EP_BIAS | EP_RES:
        gemm_wmma_kernel<EP_BIAS | EP_RES><<<grid, blk, 0, stream>>>(
            A, Wt, M, N, K, lda, ldw, ldc, o32, o16, bias, res, mask, scale);
        break;
      case EP_ELU1 | EP_MASK:
        gemm_wmma_kernel<EP_ELU1 | EP_MASK><<<grid, blk, 0, stream>>>(
            A, Wt, M, N, K, lda, ldw, ldc, o32, o16, bias, res, mask, scale);
        break;
      case EP_MASK:
        gemm_wmma_kernel<EP_MASK><<<grid, blk, 0, stream>>>(
            A, Wt, M, N, K, lda, ldw, ldc, o32, o16, bias, res, mask, scale);
        break;
      case EP_RELU:
        gemm_wmma_kernel<EP_RELU><<<grid, blk, 0, stream>>>(
            A, Wt, M, N, K, lda, ldw, ldc, o32, o16, bias, res, mask, scale);
        break;
      default:
        gemm_wmma_kernel<0><<<grid, blk, 0, stream>>>(
            A, Wt, M, N, K, lda, ldw, ldc, o32, o16, bias, res, mask, scale);
        break;
    }
  };

  // weight transpose+convert
  transpose_w_kernel<<<(D_ * D_ + 255) / 256, blk, 0, stream>>>(pos_w2, wt_pw2, D_, D_);
  transpose_w_kernel<<<(D_ * D_ + 255) / 256, blk, 0, stream>>>(wq, wt_q, D_, D_);
  transpose_w_kernel<<<(D_ * D_ + 255) / 256, blk, 0, stream>>>(wk, wt_k, D_, D_);
  transpose_w_kernel<<<(D_ * D_ + 255) / 256, blk, 0, stream>>>(wv, wt_v, D_, D_);
  transpose_w_kernel<<<(D_ * D_ + 255) / 256, blk, 0, stream>>>(w_merge, wt_m, D_, D_);
  transpose_w_kernel<<<(4 * D_ * D_ + 255) / 256, blk, 0, stream>>>(mlp_w1, wt_m1, 2 * D_, 2 * D_);
  transpose_w_kernel<<<(2 * D_ * D_ + 255) / 256, blk, 0, stream>>>(mlp_w2, wt_m2, 2 * D_, D_);
  convert_f16_kernel<<<(MT * D_ + 255) / 256, blk, 0, stream>>>(template_feat, tmpl_h, MT * D_);

  // pos MLP layer 1 (K=3, VALU) then layer 2 (WMMA) with bias + residual
  pos_hidden_kernel<<<(MS * D_ + 255) / 256, blk, 0, stream>>>(search_coors, pos_w1, pos_b1, hid0_h, MS);
  gemm(EP_BIAS | EP_RES, hid0_h, wt_pw2, MS, D_, D_, D_, D_, D_, nullptr,
       sfp_h, pos_b2, search_feat, nullptr, 1.0f);

  // Q/K/V projections with elu+1 / mask epilogues
  gemm(EP_ELU1 | EP_MASK, sfp_h, wt_q, MS, D_, D_, D_, D_, D_, nullptr, Qf,
       nullptr, nullptr, smask, 1.0f);
  gemm(EP_ELU1 | EP_MASK, tmpl_h, wt_k, MT, D_, D_, D_, D_, D_, nullptr, Kf,
       nullptr, nullptr, tmask, 1.0f);
  gemm(EP_MASK, tmpl_h, wt_v, MT, D_, D_, D_, D_, D_, nullptr, Vf, nullptr,
       nullptr, tmask, 1.0f / (float)LT_);

  // linear attention pieces
  ksum_kernel<<<(B_ * D_ + 255) / 256, blk, 0, stream>>>(Kf, Ksum);
  kv_kernel<<<B_ * H_, dim3(128), 0, stream>>>(Kf, Vf, KVt);
  z_kernel<<<(MS * H_ + 255) / 256, blk, 0, stream>>>(Qf, Ksum, Zbuf);
  attn_msg_kernel<<<B_ * H_ * (LS_ / 32) / 8, blk, 0, stream>>>(Qf, KVt, Zbuf, msg_h);

  // merge heads, LN1 + concat, MLP, LN2 + residual
  gemm(0, msg_h, wt_m, MS, D_, D_, D_, D_, D_, msgM, nullptr, nullptr, nullptr,
       nullptr, 1.0f);
  ln1_cat_kernel<<<MS / 8, blk, 0, stream>>>(msgM, search_feat, ln1_g, ln1_b, cat_h);
  gemm(EP_RELU, cat_h, wt_m1, MS, 2 * D_, 2 * D_, 2 * D_, 2 * D_, 2 * D_,
       nullptr, hid_h, nullptr, nullptr, nullptr, 1.0f);
  gemm(0, hid_h, wt_m2, MS, D_, 2 * D_, 2 * D_, 2 * D_, D_, mlpo, nullptr,
       nullptr, nullptr, nullptr, 1.0f);
  ln2_res_kernel<<<MS / 8, blk, 0, stream>>>(mlpo, search_feat, ln2_g, ln2_b, outp);
}
